// TrellisNet_13065290515031
// MI455X (gfx1250) — matrix operand order repaired and checked by reading, not compiled, not run on hardware
//
#include <hip/hip_runtime.h>
#include <hip/hip_bf16.h>

// TrellisNet on MI455X (gfx1250): 40 dependent GEMMs (M=4000,K=2000,N=2048)
// in f16 WMMA with f32 accumulate + LSTM pointwise epilogue per level.
// ~1.33 TFLOP total; whole working set (~110MB) resides in the 192MB L2,
// so the GEMM is compute-bound -> v_wmma_f32_16x16x32_f16 path.
// Staging uses CDNA5 async copies (global_load_async_to_lds_b128, ASYNCcnt)
// with a double-buffered BK=64 LDS pipeline: 16 WMMAs per barrier.

typedef __attribute__((ext_vector_type(16))) _Float16 v16h;
typedef __attribute__((ext_vector_type(8)))  _Float16 v8h;
typedef __attribute__((ext_vector_type(8)))  float    v8f;

#define NINP 400
#define HS   1000
#define NOUT 400
#define BT   2048       // B*T = 8*256
#define TT   256
#define MPAD 4096       // 4*HS = 4000 padded
#define K1   1024       // 2*512 (NINP=400 padded to 512 per tap)
#define K2   2048       // 2*1024 (HS=1000 padded to 1024 per tap)

#define BM 128
#define BN 128
#define BK 64

#define USE_ASYNC_LDS 1

#if USE_ASYNC_LDS
// copy 16B global -> LDS via the async data mover (no VGPR staging).
// vdst field = per-lane LDS byte offset (flat LDS addr truncates to offset),
// vaddr = 64-bit global address. Tracked by ASYNCcnt.
#define ASYNC_CP16(ldsp, gp)                                                 \
  asm volatile("global_load_async_to_lds_b128 %0, %1, off"                   \
               :: "v"((unsigned)(size_t)(ldsp)), "v"(gp) : "memory")
#define ASYNC_WAIT() asm volatile("s_wait_asynccnt 0" ::: "memory")
#else
#define ASYNC_CP16(ldsp, gp) (*(v8h*)(ldsp) = *(const v8h*)(gp))
#define ASYNC_WAIT()
#endif

// ---------------------------------------------------------------- pack w1/w2
// A[g][Ktot] f16 row-major, Ktot = 2*Khalf. tap0 in [0,Khalf), tap1 in [Khalf,2*Khalf)
__global__ __launch_bounds__(256) void pack_w(const float* __restrict__ w,
                                              _Float16* __restrict__ A,
                                              int Cin, int Khalf, int Ktot) {
  long long idx = (long long)blockIdx.x * 256 + threadIdx.x;
  long long tot = (long long)MPAD * Ktot;
  if (idx >= tot) return;
  int g  = (int)(idx / Ktot);
  int kk = (int)(idx % Ktot);
  int tap = (kk >= Khalf) ? 1 : 0;
  int c   = kk - tap * Khalf;
  float v = 0.f;
  if (g < 4 * HS && c < Cin) v = w[((long long)g * Cin + c) * 2 + tap];
  A[idx] = (_Float16)v;
}

// ---------------------------------------------------------------- pack X (N-major B)
// BX[col][kk], col = b*256+t, kk<512: X[b,c,t-1] (0 at t==0); kk>=512: X[b,c,t]
__global__ __launch_bounds__(256) void pack_X(const float* __restrict__ X,
                                              _Float16* __restrict__ BX) {
  long long idx = (long long)blockIdx.x * 256 + threadIdx.x;
  if (idx >= (long long)BT * K1) return;
  int col = (int)(idx >> 10);
  int kk  = (int)(idx & 1023);
  int b = col >> 8, t = col & 255;
  float v = 0.f;
  if (kk < 512) {
    int c = kk;
    if (c < NINP && t > 0) v = X[((long long)b * NINP + c) * TT + t - 1];
  } else {
    int c = kk - 512;
    if (c < NINP) v = X[((long long)b * NINP + c) * TT + t];
  }
  BX[idx] = (_Float16)v;
}

// ---------------------------------------------------------------- init Bact (N-major)
// Level-0 activations: ht==0 everywhere; left-pad column (t==0, tap0) = hid.
__global__ __launch_bounds__(256) void init_Bact(const float* __restrict__ hid,
                                                 _Float16* __restrict__ Bact) {
  long long idx = (long long)blockIdx.x * 256 + threadIdx.x;
  if (idx >= (long long)BT * K2) return;
  int col = (int)(idx >> 11);
  int kk  = (int)(idx & 2047);
  int b = col >> 8, t = col & 255;
  float v = 0.f;
  if (kk < 1024) {
    int c = kk;
    if (c < HS && t == 0) v = hid[b * HS + c];
  }
  Bact[idx] = (_Float16)v;
}

__global__ __launch_bounds__(256) void fill0(float* __restrict__ p, long long n) {
  long long idx = (long long)blockIdx.x * 256 + threadIdx.x;
  if (idx < n) p[idx] = 0.f;
}

// ---------------------------------------------------------------- WMMA GEMM
// C[MPAD][BT] (f32) = A[MPAD][K] (f16) x Bn[BT][K] (f16, N-major) + init
// init: Cpre tile if Cpre!=null, else bias[g] (guarded), else 0.
// Block tile 128x128, 8 waves (32x64 per wave = 8 v8f accumulators),
// BK=64 double-buffered LDS filled by async copies: 16 WMMA per barrier.
__global__ __launch_bounds__(256) void gemm_wmma(const _Float16* __restrict__ A,
                                                 const _Float16* __restrict__ Bn,
                                                 const float* __restrict__ Cpre,
                                                 const float* __restrict__ bias,
                                                 float* __restrict__ Cout,
                                                 int K) {
  __shared__ __align__(32) _Float16 As[2][BM][BK];   // 2 x 16 KB
  __shared__ __align__(32) _Float16 Bs[2][BN][BK];   // 2 x 16 KB

  const int tid  = threadIdx.x;
  const int wave = tid >> 5;
  const int lane = tid & 31;
  const int m0 = blockIdx.y * BM;
  const int n0 = blockIdx.x * BN;
  const int wm = (wave & 3) * 32;   // 4 waves tile M
  const int wn = (wave >> 2) * 64;  // 2 waves tile N
  const int l15 = lane & 15;
  const bool hi = lane >= 16;

  v8f acc[2][4];
#pragma unroll
  for (int mi = 0; mi < 2; ++mi)
#pragma unroll
    for (int ni = 0; ni < 4; ++ni) {
      const int n  = n0 + wn + ni * 16 + l15;
      const int mb = m0 + wm + mi * 16 + (hi ? 8 : 0);
      v8f c;
      if (Cpre) {
#pragma unroll
        for (int r = 0; r < 8; ++r) c[r] = Cpre[(long long)(mb + r) * BT + n];
      } else {
#pragma unroll
        for (int r = 0; r < 8; ++r) {
          int g = mb + r;
          c[r] = (bias && g < 4 * HS) ? bias[g] : 0.f;
        }
      }
      acc[mi][ni] = c;
    }

  // staging map: 256 threads x (4+4) x 16B = 32 KB per stage
  const int lr = tid >> 1;            // 0..127 (tile row / tile col)
  const int lc = (tid & 1) * 32;      // half-row offset in halves

#define STAGE(k0, buf)                                                        \
  {                                                                           \
    _Pragma("unroll")                                                         \
    for (int j = 0; j < 4; ++j) {                                             \
      ASYNC_CP16(&As[buf][lr][lc + j * 8],                                    \
                 &A[(long long)(m0 + lr) * K + (k0) + lc + j * 8]);           \
      ASYNC_CP16(&Bs[buf][lr][lc + j * 8],                                    \
                 &Bn[(long long)(n0 + lr) * K + (k0) + lc + j * 8]);          \
    }                                                                         \
  }

  STAGE(0, 0);
  ASYNC_WAIT();
  __syncthreads();

  const int NS = K / BK;
  for (int s = 0; s < NS; ++s) {
    const int buf = s & 1;
    if (s + 1 < NS) STAGE((s + 1) * BK, buf ^ 1);

#pragma unroll
    for (int ks = 0; ks < BK; ks += 32) {
      // A fragment: lanes 0-15 K 0..15; lanes 16-31 K 16..31
      const int khalf = ks + (hi ? 16 : 0);
      v16h afrag[2];
#pragma unroll
      for (int mi = 0; mi < 2; ++mi) {
        const int row = wm + mi * 16 + l15;
        v8h lo = *(const v8h*)&As[buf][row][khalf];
        v8h hh = *(const v8h*)&As[buf][row][khalf + 8];
        afrag[mi] = __builtin_shufflevector(lo, hh, 0, 1, 2, 3, 4, 5, 6, 7,
                                                    8, 9, 10, 11, 12, 13, 14, 15);
      }
      // B fragment: col = lane&15; lanes 0-15 hold K 0..15, lanes 16-31 K 16..31
      const int kb = ks + (hi ? 16 : 0);
      v16h bfrag[4];
#pragma unroll
      for (int ni = 0; ni < 4; ++ni) {
        const int coln = wn + ni * 16 + l15;
        bfrag[ni] = *(const v16h*)&Bs[buf][coln][kb];
      }
#pragma unroll
      for (int mi = 0; mi < 2; ++mi)
#pragma unroll
        for (int ni = 0; ni < 4; ++ni)
          acc[mi][ni] = __builtin_amdgcn_wmma_f32_16x16x32_f16(
              false, afrag[mi], false, bfrag[ni], (short)0, acc[mi][ni],
              false, false);
    }
    ASYNC_WAIT();
    __syncthreads();
  }

#pragma unroll
  for (int mi = 0; mi < 2; ++mi)
#pragma unroll
    for (int ni = 0; ni < 4; ++ni) {
      const int n  = n0 + wn + ni * 16 + l15;
      const int mb = m0 + wm + mi * 16 + (hi ? 8 : 0);
      v8f c = acc[mi][ni];
#pragma unroll
      for (int r = 0; r < 8; ++r) Cout[(long long)(mb + r) * BT + n] = c[r];
    }
}

// ---------------------------------------------------------------- LSTM pointwise
__global__ __launch_bounds__(256) void lstm_pointwise(
    const float* __restrict__ outg,     // [MPAD][BT]
    const float* __restrict__ ct_prev,  // [HS][BT]
    float* __restrict__ ct_next,        // [HS][BT]
    _Float16* __restrict__ Bact,        // [BT][K2] N-major, next level activations
    const float* __restrict__ hid,      // [8][HS]
    const float* __restrict__ cell,     // [8][HS]
    float* __restrict__ out_main, float* __restrict__ ht_last,
    float* __restrict__ ct_last, float* __restrict__ aux, int level) {
  long long idx = (long long)blockIdx.x * 256 + threadIdx.x;
  if (idx >= (long long)HS * BT) return;
  const int c   = (int)(idx >> 11);
  const int col = (int)(idx & 2047);
  const int b = col >> 8, t = col & 255;

  const float oi = outg[(long long)(0 * HS + c) * BT + col];
  const float oo = outg[(long long)(1 * HS + c) * BT + col];
  const float og = outg[(long long)(2 * HS + c) * BT + col];
  const float of = outg[(long long)(3 * HS + c) * BT + col];

  const float it = 1.f / (1.f + __expf(-oi));
  const float ot = 1.f / (1.f + __expf(-oo));
  const float gt = tanhf(og);
  const float ft = 1.f / (1.f + __expf(-of));

  const float c1 = (t == 0) ? cell[b * HS + c]
                            : ct_prev[(long long)c * BT + col - 1];
  const float ct = ft * c1 + it * gt;
  const float ht = ot * tanhf(ct);

  ct_next[(long long)c * BT + col] = ct;

  const _Float16 h16 = (_Float16)ht;
  Bact[(long long)col * K2 + 1024 + c] = h16;              // tap1: ht[c,t]
  if (t < 255) Bact[(long long)(col + 1) * K2 + c] = h16;  // tap0: ht[c,t] -> t+1
  if (t == 0)  Bact[(long long)col * K2 + c] = (_Float16)hid[b * HS + c];

  if ((level % 20) == 19 && c >= HS - NOUT) {
    const int ai = level / 20;  // 0 or 1
    aux[((long long)(b * 2 + ai) * TT + t) * NOUT + (c - (HS - NOUT))] = ht;
  }
  if (level == 39) {
    if (c >= HS - NOUT)
      out_main[((long long)b * TT + t) * NOUT + (c - (HS - NOUT))] = ht;
    if (t == 255) {
      ht_last[b * HS + c] = ht;
      ct_last[b * HS + c] = ct;
    }
  }
}

// ---------------------------------------------------------------- launch
extern "C" void kernel_launch(void* const* d_in, const int* in_sizes, int n_in,
                              void* d_out, int out_size, void* d_ws, size_t ws_size,
                              hipStream_t stream) {
  const float* X    = (const float*)d_in[0];
  const float* hid  = (const float*)d_in[1];
  const float* cell = (const float*)d_in[2];
  const float* w1   = (const float*)d_in[3];
  const float* w2   = (const float*)d_in[4];
  const float* b2   = (const float*)d_in[5];

  char* ws = (char*)d_ws;
  size_t o = 0;
  _Float16* A1   = (_Float16*)(ws + o); o += (size_t)MPAD * K1 * 2;   //  8.4 MB
  _Float16* A2   = (_Float16*)(ws + o); o += (size_t)MPAD * K2 * 2;   // 16.8 MB
  _Float16* BX   = (_Float16*)(ws + o); o += (size_t)BT * K1 * 2;     //  4.2 MB
  _Float16* Bact = (_Float16*)(ws + o); o += (size_t)BT * K2 * 2;     //  8.4 MB
  float*    pre  = (float*)(ws + o);    o += (size_t)MPAD * BT * 4;   // 33.6 MB
  float*    outg = (float*)(ws + o);    o += (size_t)MPAD * BT * 4;   // 33.6 MB
  float*    ctA  = (float*)(ws + o);    o += (size_t)HS * BT * 4;     //  8.2 MB
  float*    ctB  = (float*)(ws + o);    o += (size_t)HS * BT * 4;     //  8.2 MB

  float* out_main = (float*)d_out;                 // (8,256,400)
  float* ht_last  = out_main + (size_t)8 * TT * NOUT;
  float* ct_last  = ht_last + (size_t)8 * HS;
  float* aux      = ct_last + (size_t)8 * HS;      // (8,2,256,400)

  const int T256 = 256;
  pack_w<<<(int)(((long long)MPAD * K1 + 255) / 256), T256, 0, stream>>>(w1, A1, NINP, 512, K1);
  pack_w<<<(int)(((long long)MPAD * K2 + 255) / 256), T256, 0, stream>>>(w2, A2, HS, 1024, K2);
  pack_X<<<(int)(((long long)BT * K1 + 255) / 256), T256, 0, stream>>>(X, BX);
  init_Bact<<<(int)(((long long)BT * K2 + 255) / 256), T256, 0, stream>>>(hid, Bact);
  fill0<<<(int)(((long long)HS * BT + 255) / 256), T256, 0, stream>>>(ctA, (long long)HS * BT);

  // pre = conv1(X) + b2 : M=4000 K=1024(padded 800) N=2048
  gemm_wmma<<<dim3(BT / BN, MPAD / BM), T256, 0, stream>>>(A1, BX, nullptr, b2, pre, K1);

  for (int lv = 0; lv < 40; ++lv) {
    gemm_wmma<<<dim3(BT / BN, MPAD / BM), T256, 0, stream>>>(A2, Bact, pre, nullptr, outg, K2);
    float* cp = (lv & 1) ? ctB : ctA;
    float* cn = (lv & 1) ? ctA : ctB;
    lstm_pointwise<<<(int)(((long long)HS * BT + 255) / 256), T256, 0, stream>>>(
        outg, cp, cn, Bact, hid, cell, out_main, ht_last, ct_last, aux, lv);
  }
}
